// LlamaQMLP_21947282883010
// MI455X (gfx1250) — matrix operand order repaired and checked by reading, not compiled, not run on hardware
//
#include <hip/hip_runtime.h>
#include <hip/hip_bf16.h>

typedef __attribute__((ext_vector_type(16))) _Float16 v16h;
typedef __attribute__((ext_vector_type(8)))  _Float16 v8h;
typedef __attribute__((ext_vector_type(4)))  _Float16 v4h;
typedef __attribute__((ext_vector_type(2)))  _Float16 v2h;
typedef __attribute__((ext_vector_type(8)))  float    v8f;

#define HID    4096
#define INTER  11008
#define GSIZE  128
#define BM     128
#define BN     64
#define BK     64
#define NKB_GU (HID / BK)
#define NKB_DN (INTER / BK)
#define NW_GU  (2 * INTER / 8)   /* 2752 int32 words per K-row of gate_up */
#define NW_DN  (HID / 8)         /* 512  int32 words per K-row of down    */
#define LDK    (BK + 8)          /* padded LDS row stride in halves       */

// Assemble a 16xf16 WMMA fragment from two 16-byte LDS chunks.
__device__ __forceinline__ v16h frag2(const _Float16* p0, const _Float16* p1) {
    v8h a0 = *(const v8h*)p0;
    v8h a1 = *(const v8h*)p1;
    v16h a;
#pragma unroll
    for (int i = 0; i < 8; ++i) { a[i] = a0[i]; a[i + 8] = a1[i]; }
    return a;
}

// ---------------------------------------------------------------------------
// Kernel 1: gate_up GEMM with fused int4 dequant + SiLU(gate)*up -> h (f16)
// 8 waves tiled 4(M) x 2(N); each wave owns 32M x 32N for gate AND up.
// Register-staged prefetch: iteration k+1's global loads issue before the
// WMMA block of iteration k, hiding VMEM latency under the matrix pipe.
// ---------------------------------------------------------------------------
__global__ __launch_bounds__(256)
void k_gateup_silu(const float* __restrict__ x,
                   const int*   __restrict__ qw,
                   const int*   __restrict__ qz,
                   const float* __restrict__ sc,
                   _Float16*    __restrict__ h) {
    __shared__ _Float16 sA [BM][LDK];
    __shared__ _Float16 sBg[BN][LDK];
    __shared__ _Float16 sBu[BN][LDK];

    const int tid  = threadIdx.x;
    const int wave = tid >> 5;
    const int lane = tid & 31;
    const int m0   = blockIdx.x * BM;
    const int n0   = blockIdx.y * BN;
    const int wm   = (wave >> 1) * 32;
    const int wn   = (wave & 1) * 32;

    const int dq_w   = tid & 7;               // word (8 cols of 8 nibbles)
    const int dq_k   = (tid >> 3) * 2;        // k pair base, 0..62
    const int dq_col = (n0 >> 3) + dq_w;

    v8f accG[2][2] = {};
    v8f accU[2][2] = {};

    // ---- register staging state for one K-tile ----
    float4   ax[8];
    unsigned wg0, wg1, wu0, wu1, zg, zu;
    float4   sg0, sg1, su0, su1;

    auto stage = [&](int kb) {                // global -> registers
        const int k0 = kb * BK;
        const int g  = k0 >> 7;
#pragma unroll
        for (int i = 0; i < 8; ++i) {
            int id = tid + i * 256;           // float4 grid: 128 rows x 16 cols
            int r  = id >> 4;
            int c4 = id & 15;
            ax[i] = *(const float4*)(x + (size_t)(m0 + r) * HID + k0 + c4 * 4);
        }
        const int* qr0 = qw + (size_t)(k0 + dq_k) * NW_GU;
        const int* qr1 = qr0 + NW_GU;
        wg0 = (unsigned)qr0[dq_col];
        wg1 = (unsigned)qr1[dq_col];
        wu0 = (unsigned)qr0[(INTER / 8) + dq_col];
        wu1 = (unsigned)qr1[(INTER / 8) + dq_col];
        zg  = (unsigned)qz[(size_t)g * NW_GU + dq_col];
        zu  = (unsigned)qz[(size_t)g * NW_GU + (INTER / 8) + dq_col];
        const float4* sgp = (const float4*)(sc + (size_t)g * (2 * INTER) + n0 + dq_w * 8);
        const float4* sup = (const float4*)(sc + (size_t)g * (2 * INTER) + INTER + n0 + dq_w * 8);
        sg0 = sgp[0]; sg1 = sgp[1];
        su0 = sup[0]; su1 = sup[1];
    };

    auto commit = [&]() {                     // registers -> LDS (cvt + dequant)
#pragma unroll
        for (int i = 0; i < 8; ++i) {
            int id = tid + i * 256;
            int r  = id >> 4;
            int c4 = id & 15;
            v4h p = { (_Float16)ax[i].x, (_Float16)ax[i].y,
                      (_Float16)ax[i].z, (_Float16)ax[i].w };
            *(v4h*)&sA[r][c4 * 4] = p;
        }
        float s_g[8] = { sg0.x, sg0.y, sg0.z, sg0.w, sg1.x, sg1.y, sg1.z, sg1.w };
        float s_u[8] = { su0.x, su0.y, su0.z, su0.w, su1.x, su1.y, su1.z, su1.w };
#pragma unroll
        for (int q = 0; q < 8; ++q) {
            int zgq = (int)((zg >> (4 * q)) & 0xF);
            int zuq = (int)((zu >> (4 * q)) & 0xF);
            int nl  = dq_w * 8 + q;
            v2h pg = { (_Float16)((float)((int)((wg0 >> (4 * q)) & 0xF) - zgq) * s_g[q]),
                       (_Float16)((float)((int)((wg1 >> (4 * q)) & 0xF) - zgq) * s_g[q]) };
            v2h pu = { (_Float16)((float)((int)((wu0 >> (4 * q)) & 0xF) - zuq) * s_u[q]),
                       (_Float16)((float)((int)((wu1 >> (4 * q)) & 0xF) - zuq) * s_u[q]) };
            *(v2h*)&sBg[nl][dq_k] = pg;
            *(v2h*)&sBu[nl][dq_k] = pu;
        }
    };

    stage(0);
    commit();
    __syncthreads();

    const int arow = wm + (lane & 15);
    const int kh   = (lane >> 4) * 8;
    const int ncol = wn + (lane & 15);
    const int klo  = (lane >> 4) * 16;

    for (int kb = 0; kb < NKB_GU; ++kb) {
        if (kb + 1 < NKB_GU) stage(kb + 1);   // prefetch next tile into regs

        // ---- WMMA: 32M x 32N register block, gate + up ----
#pragma unroll
        for (int kk = 0; kk < BK; kk += 32) {
            v16h a[2];
#pragma unroll
            for (int mi = 0; mi < 2; ++mi)
                a[mi] = frag2(&sA[arow + mi * 16][kk + kh],
                              &sA[arow + mi * 16][kk + 16 + kh]);
#pragma unroll
            for (int ni = 0; ni < 2; ++ni) {
                v16h bg = frag2(&sBg[ncol + ni * 16][kk + kh],
                                &sBg[ncol + ni * 16][kk + 16 + kh]);
#pragma unroll
                for (int mi = 0; mi < 2; ++mi)
                    accG[mi][ni] = __builtin_amdgcn_wmma_f32_16x16x32_f16(
                        false, a[mi], false, bg, (short)0, accG[mi][ni], false, false);
                v16h bu = frag2(&sBu[ncol + ni * 16][kk + kh],
                                &sBu[ncol + ni * 16][kk + 16 + kh]);
#pragma unroll
                for (int mi = 0; mi < 2; ++mi)
                    accU[mi][ni] = __builtin_amdgcn_wmma_f32_16x16x32_f16(
                        false, a[mi], false, bu, (short)0, accU[mi][ni], false, false);
            }
        }
        __syncthreads();                      // all waves done reading LDS
        if (kb + 1 < NKB_GU) commit();        // write next tile
        __syncthreads();                      // next tile visible
    }

    // ---- epilogue: h = silu(gate) * up, stored f16 ----
    const int mb = m0 + wm + 8 * (lane >> 4);
    const int nb = n0 + wn + (lane & 15);
#pragma unroll
    for (int mi = 0; mi < 2; ++mi) {
#pragma unroll
        for (int ni = 0; ni < 2; ++ni) {
#pragma unroll
            for (int r = 0; r < 8; ++r) {
                float gv = accG[mi][ni][r];
                float uv = accU[mi][ni][r];
                float hv = gv / (1.0f + __expf(-gv)) * uv;
                h[(size_t)(mb + mi * 16 + r) * INTER + nb + ni * 16] = (_Float16)hv;
            }
        }
    }
}

// ---------------------------------------------------------------------------
// Kernel 2: down GEMM with fused int4 dequant, same pipelining scheme.
// ---------------------------------------------------------------------------
__global__ __launch_bounds__(256)
void k_down(const _Float16* __restrict__ h,
            const int*      __restrict__ qw,
            const int*      __restrict__ qz,
            const float*    __restrict__ sc,
            float*          __restrict__ out) {
    __shared__ _Float16 sA[BM][LDK];
    __shared__ _Float16 sB[BN][LDK];

    const int tid  = threadIdx.x;
    const int wave = tid >> 5;
    const int lane = tid & 31;
    const int m0   = blockIdx.x * BM;
    const int n0   = blockIdx.y * BN;
    const int wm   = (wave >> 1) * 32;
    const int wn   = (wave & 1) * 32;

    const int dq_w   = tid & 7;
    const int dq_k   = (tid >> 3) * 2;
    const int dq_col = (n0 >> 3) + dq_w;

    v8f acc[2][2] = {};

    v8h      axh[8];
    unsigned w0, w1, zv;
    float4   s0, s1;

    auto stage = [&](int kb) {
        const int k0 = kb * BK;
        const int g  = k0 >> 7;
#pragma unroll
        for (int i = 0; i < 8; ++i) {
            int id = tid + i * 256;           // v8h grid: 128 rows x 8 cols
            int r  = id >> 3;
            int c8 = id & 7;
            axh[i] = *(const v8h*)(h + (size_t)(m0 + r) * INTER + k0 + c8 * 8);
        }
        const int* qr0 = qw + (size_t)(k0 + dq_k) * NW_DN;
        const int* qr1 = qr0 + NW_DN;
        w0 = (unsigned)qr0[dq_col];
        w1 = (unsigned)qr1[dq_col];
        zv = (unsigned)qz[(size_t)g * NW_DN + dq_col];
        const float4* sp = (const float4*)(sc + (size_t)g * HID + n0 + dq_w * 8);
        s0 = sp[0]; s1 = sp[1];
    };

    auto commit = [&]() {
#pragma unroll
        for (int i = 0; i < 8; ++i) {
            int id = tid + i * 256;
            int r  = id >> 3;
            int c8 = id & 7;
            *(v8h*)&sA[r][c8 * 8] = axh[i];
        }
        float s[8] = { s0.x, s0.y, s0.z, s0.w, s1.x, s1.y, s1.z, s1.w };
#pragma unroll
        for (int q = 0; q < 8; ++q) {
            int zq = (int)((zv >> (4 * q)) & 0xF);
            v2h p = { (_Float16)((float)((int)((w0 >> (4 * q)) & 0xF) - zq) * s[q]),
                      (_Float16)((float)((int)((w1 >> (4 * q)) & 0xF) - zq) * s[q]) };
            *(v2h*)&sB[dq_w * 8 + q][dq_k] = p;
        }
    };

    stage(0);
    commit();
    __syncthreads();

    const int arow = wm + (lane & 15);
    const int kh   = (lane >> 4) * 8;
    const int ncol = wn + (lane & 15);
    const int klo  = (lane >> 4) * 16;

    for (int kb = 0; kb < NKB_DN; ++kb) {     // 172 iterations
        if (kb + 1 < NKB_DN) stage(kb + 1);

#pragma unroll
        for (int kk = 0; kk < BK; kk += 32) {
            v16h a[2];
#pragma unroll
            for (int mi = 0; mi < 2; ++mi)
                a[mi] = frag2(&sA[arow + mi * 16][kk + kh],
                              &sA[arow + mi * 16][kk + 16 + kh]);
#pragma unroll
            for (int ni = 0; ni < 2; ++ni) {
                v16h b = frag2(&sB[ncol + ni * 16][kk + kh],
                               &sB[ncol + ni * 16][kk + 16 + kh]);
#pragma unroll
                for (int mi = 0; mi < 2; ++mi)
                    acc[mi][ni] = __builtin_amdgcn_wmma_f32_16x16x32_f16(
                        false, a[mi], false, b, (short)0, acc[mi][ni], false, false);
            }
        }
        __syncthreads();
        if (kb + 1 < NKB_DN) commit();
        __syncthreads();
    }

    const int mb = m0 + wm + 8 * (lane >> 4);
    const int nb = n0 + wn + (lane & 15);
#pragma unroll
    for (int mi = 0; mi < 2; ++mi) {
#pragma unroll
        for (int ni = 0; ni < 2; ++ni) {
#pragma unroll
            for (int r = 0; r < 8; ++r) {
                out[(size_t)(mb + mi * 16 + r) * HID + nb + ni * 16] = acc[mi][ni][r];
            }
        }
    }
}

extern "C" void kernel_launch(void* const* d_in, const int* in_sizes, int n_in,
                              void* d_out, int out_size, void* d_ws, size_t ws_size,
                              hipStream_t stream) {
    const float* x     = (const float*)d_in[0];
    const int*   qw_gu = (const int*)  d_in[1];
    const int*   qz_gu = (const int*)  d_in[2];
    const float* sc_gu = (const float*)d_in[3];
    const int*   qw_dn = (const int*)  d_in[4];
    const int*   qz_dn = (const int*)  d_in[5];
    const float* sc_dn = (const float*)d_in[6];
    float*       out   = (float*)d_out;
    _Float16*    hbuf  = (_Float16*)d_ws;      // [M, INTER] f16 intermediate

    const int M = in_sizes[0] / HID;           // 4096 tokens

    dim3 g1(M / BM, INTER / BN);               // 32 x 172
    k_gateup_silu<<<g1, 256, 0, stream>>>(x, qw_gu, qz_gu, sc_gu, hbuf);

    dim3 g2(M / BM, HID / BN);                 // 32 x 64
    k_down<<<g2, 256, 0, stream>>>(hbuf, qw_dn, qz_dn, sc_dn, out);
}
